// FP8Linear_11098195493488
// MI455X (gfx1250) — 8-bit matrix operand order repaired and checked by reading, compiled, not run on hardware
//
#include <hip/hip_runtime.h>
#include <hip/hip_bf16.h>
#include <stdint.h>

#define KDIM   4096
#define NDIM   4096
#define KSTEP  128
#define KSTEPS (KDIM / KSTEP)
#define BM     128
#define BN     128
#define LDSPAD 144
#define LDSBUF (BN * LDSPAD)

#define USE_ASYNC_LDS 1

typedef int   v16i __attribute__((ext_vector_type(16)));
typedef float v8f  __attribute__((ext_vector_type(8)));
typedef unsigned int u32x2 __attribute__((ext_vector_type(2)));
typedef unsigned int u32x4 __attribute__((ext_vector_type(4)));
typedef float f32x4 __attribute__((ext_vector_type(4)));

#if !__has_builtin(__builtin_amdgcn_cvt_pk_fp8_f32)
static __device__ __forceinline__ unsigned sw_cvt_e4m3(float f) {
  unsigned bits = __float_as_uint(f);
  unsigned sign = (bits >> 24) & 0x80u;
  float a = __uint_as_float(bits & 0x7fffffffu);
  if (a >= 0.015625f) {
    int e; float m = frexpf(a, &e);
    int mi = (int)rintf(m * 16.0f);
    if (mi == 16) { mi = 8; e += 1; }
    int ecode = (e - 1) + 7;
    return sign | (unsigned)((ecode << 3) | (mi - 8));
  } else {
    int d = (int)rintf(a * 512.0f);
    return sign | (unsigned)d;
  }
}
#endif

static __device__ __forceinline__ unsigned pack4_e4m3(float v0, float v1, float v2, float v3) {
#if __has_builtin(__builtin_amdgcn_cvt_pk_fp8_f32)
  int p = __builtin_amdgcn_cvt_pk_fp8_f32(v0, v1, 0, false);
  p     = __builtin_amdgcn_cvt_pk_fp8_f32(v2, v3, p, true);
  return (unsigned)p;
#else
  return sw_cvt_e4m3(v0) | (sw_cvt_e4m3(v1) << 8) |
         (sw_cvt_e4m3(v2) << 16) | (sw_cvt_e4m3(v3) << 24);
#endif
}

static __device__ __forceinline__ unsigned lds_off32(void* p) {
  return (unsigned)(unsigned long long)(__attribute__((address_space(3))) void*)p;
}

#if USE_ASYNC_LDS
#define STAGE_B(gsrc, ldst) do {                                                               \
    unsigned _d = lds_off32((void*)(ldst));                                                    \
    const uint8_t* _g = (gsrc);                                                                \
    asm volatile("global_load_async_to_lds_b128 %0, %1, off"           :: "v"(_d), "v"(_g) : "memory"); \
    asm volatile("global_load_async_to_lds_b128 %0, %1, off offset:16" :: "v"(_d), "v"(_g) : "memory"); \
    asm volatile("global_load_async_to_lds_b128 %0, %1, off offset:32" :: "v"(_d), "v"(_g) : "memory"); \
    asm volatile("global_load_async_to_lds_b128 %0, %1, off offset:48" :: "v"(_d), "v"(_g) : "memory"); \
  } while (0)
#if __has_builtin(__builtin_amdgcn_s_wait_asynccnt)
#define WAIT_STAGE() __builtin_amdgcn_s_wait_asynccnt(0)
#else
#define WAIT_STAGE() asm volatile("s_wait_asynccnt 0x0" ::: "memory")
#endif
#else
#define STAGE_B(gsrc, ldst) do {                                                               \
    const u32x4* _g = (const u32x4*)(gsrc);                                                    \
    u32x4* _l = (u32x4*)(ldst);                                                                \
    u32x4 _r0 = _g[0], _r1 = _g[1], _r2 = _g[2], _r3 = _g[3];                                  \
    _l[0] = _r0; _l[1] = _r1; _l[2] = _r2; _l[3] = _r3;                                        \
  } while (0)
#define WAIT_STAGE() do {} while (0)
#endif

__global__ __launch_bounds__(256) void quant_rows(const float* __restrict__ x,
                                                  uint8_t* __restrict__ xq,
                                                  float* __restrict__ in_scale) {
  const int row = blockIdx.x;
  const int tid = threadIdx.x;
  const float* xr = x + (size_t)row * KDIM;
  const f32x4* xv = (const f32x4*)(xr + tid * 16);

  f32x4 v[4];
  float amax = 0.0f;
#pragma unroll
  for (int i = 0; i < 4; ++i) {
    v[i] = xv[i];
#pragma unroll
    for (int j = 0; j < 4; ++j) amax = fmaxf(amax, fabsf(v[i][j]));
  }
#pragma unroll
  for (int off = 16; off > 0; off >>= 1)
    amax = fmaxf(amax, __shfl_xor(amax, off, 32));
  __shared__ float red[8];
  const int wv = tid >> 5, ln = tid & 31;
  if (ln == 0) red[wv] = amax;
  __syncthreads();
#pragma unroll
  for (int i = 0; i < 8; ++i) amax = fmaxf(amax, red[i]);

  const float s = fmaxf(amax * (1.0f / 448.0f), 1e-12f);
  if (tid == 0) in_scale[row] = s;
  const float inv = 1.0f / s;

  unsigned q[4];
#pragma unroll
  for (int i = 0; i < 4; ++i) {
    float c0 = fminf(fmaxf(v[i][0] * inv, -448.0f), 448.0f);
    float c1 = fminf(fmaxf(v[i][1] * inv, -448.0f), 448.0f);
    float c2 = fminf(fmaxf(v[i][2] * inv, -448.0f), 448.0f);
    float c3 = fminf(fmaxf(v[i][3] * inv, -448.0f), 448.0f);
    q[i] = pack4_e4m3(c0, c1, c2, c3);
  }

  const int m = row & 15;
  const int kb = tid >> 3, j = tid & 7;
  const size_t base = ((size_t)(row >> 4) * KSTEPS + kb) * 2048;
  u32x4 qv = { q[0], q[1], q[2], q[3] };
  *(u32x4*)(xq + base + (size_t)((j & 1) * 16 + m) * 64 + (j >> 1) * 16) = qv;
}

static __device__ __forceinline__ v16i load_btile(const uint8_t* rb, int t) {
  v16i B;
#pragma unroll
  for (int g = 0; g < 4; ++g) {
    u32x4 d = *(const u32x4*)(rb + t * (16 * LDSPAD) + g * 32);
#pragma unroll
    for (int q = 0; q < 4; ++q) B[4 * g + q] = (int)d[q];
  }
  return B;
}

static __device__ __forceinline__ void compute_step(
    const uint8_t* a0p, const uint8_t* a1p, const uint8_t* rb,
    v8f& c00, v8f& c01, v8f& c02, v8f& c03,
    v8f& c10, v8f& c11, v8f& c12, v8f& c13) {
  const u32x4* a0 = (const u32x4*)a0p;
  const u32x4* a1 = (const u32x4*)a1p;
  v16i A0, A1;
#pragma unroll
  for (int c = 0; c < 4; ++c) {
    u32x4 d0 = a0[c], d1 = a1[c];
#pragma unroll
    for (int q = 0; q < 4; ++q) { A0[4 * c + q] = (int)d0[q]; A1[4 * c + q] = (int)d1[q]; }
  }
  v16i Ba = load_btile(rb, 0);
  v16i Bb = load_btile(rb, 1);
  c00 = __builtin_amdgcn_wmma_f32_16x16x128_fp8_fp8(A0, Ba, (short)0, c00, false, false);
  c10 = __builtin_amdgcn_wmma_f32_16x16x128_fp8_fp8(A1, Ba, (short)0, c10, false, false);
  Ba = load_btile(rb, 2);
  c01 = __builtin_amdgcn_wmma_f32_16x16x128_fp8_fp8(A0, Bb, (short)0, c01, false, false);
  c11 = __builtin_amdgcn_wmma_f32_16x16x128_fp8_fp8(A1, Bb, (short)0, c11, false, false);
  Bb = load_btile(rb, 3);
  c02 = __builtin_amdgcn_wmma_f32_16x16x128_fp8_fp8(A0, Ba, (short)0, c02, false, false);
  c12 = __builtin_amdgcn_wmma_f32_16x16x128_fp8_fp8(A1, Ba, (short)0, c12, false, false);
  c03 = __builtin_amdgcn_wmma_f32_16x16x128_fp8_fp8(A0, Bb, (short)0, c03, false, false);
  c13 = __builtin_amdgcn_wmma_f32_16x16x128_fp8_fp8(A1, Bb, (short)0, c13, false, false);
}

__global__ __launch_bounds__(256) void fp8_gemm(const uint8_t* __restrict__ xq,
                                                const uint8_t* __restrict__ w,
                                                const float* __restrict__ in_scale,
                                                const float* __restrict__ w_scale,
                                                const float* __restrict__ bias,
                                                float* __restrict__ out) {
  __shared__ __attribute__((aligned(16))) uint8_t ldsB[2 * LDSBUF];

  const int tid  = threadIdx.x;
  const int wave = tid >> 5;
  const int lane = tid & 31;
  const int lo = lane & 15, hi = lane >> 4;
  const int mw = wave >> 1, nw = wave & 1;

  const int mBlk  = blockIdx.y * BM;
  const int nBlk  = blockIdx.x * BN;
  const int mBase = mBlk + mw * 32;
  const int nBase = nBlk + nw * 64;

  const uint8_t* aPtr0 = xq + (size_t)(mBase >> 4) * KSTEPS * 2048 + lane * 64;
  const uint8_t* aPtr1 = aPtr0 + (size_t)KSTEPS * 2048;

  const int scol = tid >> 1, shalf = tid & 1;
  const uint8_t* wPtr = w + (size_t)(nBlk + scol) * KDIM + shalf * 64;
  uint8_t* ldsWr0 = ldsB + scol * LDSPAD + shalf * 64;
  uint8_t* ldsWr1 = ldsWr0 + LDSBUF;

  const uint8_t* ldsRd0 = ldsB + (nw * 64 + lo) * LDSPAD + hi * 16;
  const uint8_t* ldsRd1 = ldsRd0 + LDSBUF;

  v8f c00 = {}, c01 = {}, c02 = {}, c03 = {};
  v8f c10 = {}, c11 = {}, c12 = {}, c13 = {};

  STAGE_B(wPtr, ldsWr0);

  for (int kb = 0; kb < KSTEPS; kb += 2) {
    WAIT_STAGE();
    __syncthreads();
    STAGE_B(wPtr + (size_t)(kb + 1) * KSTEP, ldsWr1);
    compute_step(aPtr0 + (size_t)kb * 2048, aPtr1 + (size_t)kb * 2048, ldsRd0,
                 c00, c01, c02, c03, c10, c11, c12, c13);

    WAIT_STAGE();
    __syncthreads();
    if (kb + 2 < KSTEPS)
      STAGE_B(wPtr + (size_t)(kb + 2) * KSTEP, ldsWr0);
    compute_step(aPtr0 + (size_t)(kb + 1) * 2048, aPtr1 + (size_t)(kb + 1) * 2048, ldsRd1,
                 c00, c01, c02, c03, c10, c11, c12, c13);
  }

  v8f accs[2][4] = { { c00, c01, c02, c03 }, { c10, c11, c12, c13 } };
#pragma unroll
  for (int tm = 0; tm < 2; ++tm) {
    const int rowB = mBase + tm * 16 + 8 * hi;
    float rs[8];
#pragma unroll
    for (int r = 0; r < 8; ++r) rs[r] = in_scale[rowB + r];
#pragma unroll
    for (int t = 0; t < 4; ++t) {
      const v8f a = accs[tm][t];
      const int col = nBase + 16 * t + lo;
      const float ws = w_scale[col];
      const float bb = bias[col];
      float* op = out + col;
#pragma unroll
      for (int r = 0; r < 8; ++r)
        op[(size_t)(rowB + r) * NDIM] = a[r] * rs[r] * ws + bb;
    }
  }
}

extern "C" void kernel_launch(void* const* d_in, const int* in_sizes, int n_in,
                              void* d_out, int out_size, void* d_ws, size_t ws_size,
                              hipStream_t stream) {
  const float*   x      = (const float*)d_in[0];
  const uint8_t* w      = (const uint8_t*)d_in[1];
  const float*   wscale = (const float*)d_in[2];
  const float*   bias   = (const float*)d_in[3];
  float*         out    = (float*)d_out;

  const int T = in_sizes[0] / KDIM;

  float*   in_scale = (float*)d_ws;
  uint8_t* xq       = (uint8_t*)d_ws + 32768;

  quant_rows<<<T, 256, 0, stream>>>(x, xq, in_scale);

  dim3 grid(NDIM / BN, T / BM);
  fp8_gemm<<<grid, 256, 0, stream>>>(xq, w, in_scale, wscale, bias, out);
}
